// ManifoldAttention_75969381531742
// MI455X (gfx1250) — matrix operand order repaired and checked by reading, not compiled, not run on hardware
//
#include <hip/hip_runtime.h>
#include <hip/hip_bf16.h>

typedef __bf16 bf16;
typedef __attribute__((ext_vector_type(16))) __bf16        v16bf;
typedef __attribute__((ext_vector_type(8)))  float         v8f;
typedef __attribute__((ext_vector_type(4)))  unsigned int  u32x4;
typedef __attribute__((ext_vector_type(8)))  int           i32x8;
typedef __attribute__((ext_vector_type(4)))  int           i32x4;

union FragA {            // 16 bf16 per lane = one WMMA A/B operand (8 VGPRs)
    v16bf v;
    uint4 q[2];
};

// Require the CDNA5 Tensor Data Mover path on the DEVICE pass only (the host
// pass never has amdgcn builtins). A successful compile now proves the device
// binary contains tensor_load_to_lds + s_wait_tensorcnt — no silent fallback.
#ifdef __HIP_DEVICE_COMPILE__
#if !__has_builtin(__builtin_amdgcn_tensor_load_to_lds)
#error "gfx1250 tensor_load_to_lds builtin not available on device pass"
#endif
#if !__has_builtin(__builtin_amdgcn_s_wait_tensorcnt)
#error "gfx1250 s_wait_tensorcnt builtin not available on device pass"
#endif
#endif

// ---------------------------------------------------------------------------
// TDM: async 2D tile DMA global -> LDS (D# built per ISA 08_async_tensor §8).
// dims/strides in 2-byte elements. One wave issues; completion via TENSORcnt.
// ---------------------------------------------------------------------------
__device__ __forceinline__ void tdm_load_2d_bf16(const void* gaddr, unsigned lds_addr,
                                                 unsigned tile_d0, unsigned tile_d1,
                                                 unsigned stride_d0,
                                                 unsigned tensor_d0, unsigned tensor_d1) {
#ifdef __HIP_DEVICE_COMPILE__
    unsigned long long ga = (unsigned long long)(uintptr_t)gaddr;
    // group0: [1:0]=count(1) | [63:32]=lds_addr | [120:64]=global_addr | [127:126]=type(2)
    u32x4 g0 = { 1u,
                 lds_addr,
                 (unsigned)(ga & 0xFFFFFFFFu),
                 ((unsigned)((ga >> 32) & 0x01FFFFFFull)) | 0x80000000u };
    // group1: [17:16]=data_size(1=2B) | [79:48]=tensor_dim0 | [111:80]=tensor_dim1
    //         [127:112]=tile_dim0 | [143:128]=tile_dim1 | [207:160]=tensor_dim0_stride
    i32x8 g1 = { (int)(1u << 16),
                 (int)((tensor_d0 & 0xFFFFu) << 16),
                 (int)((tensor_d0 >> 16) | ((tensor_d1 & 0xFFFFu) << 16)),
                 (int)((tensor_d1 >> 16) | ((tile_d0 & 0xFFFFu) << 16)),
                 (int)(tile_d1 & 0xFFFFu),
                 (int)stride_d0,
                 0, 0 };
    i32x4 g2 = { 0, 0, 0, 0 };
    i32x4 g3 = { 0, 0, 0, 0 };
#if __clang_major__ >= 23
    i32x8 g4 = { 0, 0, 0, 0, 0, 0, 0, 0 };
    __builtin_amdgcn_tensor_load_to_lds(g0, g1, g2, g3, g4, 0);
#else
    __builtin_amdgcn_tensor_load_to_lds(g0, g1, g2, g3, 0);
#endif
#endif
}

__device__ __forceinline__ void tdm_wait0() {
#ifdef __HIP_DEVICE_COMPILE__
    __builtin_amdgcn_s_wait_tensorcnt(0);
#endif
}

// ---------------------------------------------------------------------------
// f32 -> bf16 conversion (vectorized 4-wide)
// ---------------------------------------------------------------------------
__global__ void cvt_f32_to_bf16(const float* __restrict__ src,
                                bf16* __restrict__ dst, int n4) {
    int i = blockIdx.x * blockDim.x + threadIdx.x;
    if (i < n4) {
        float4 f = reinterpret_cast<const float4*>(src)[i];
        union { bf16 e[4]; uint2 u; } t;
        t.e[0] = (bf16)f.x; t.e[1] = (bf16)f.y;
        t.e[2] = (bf16)f.z; t.e[3] = (bf16)f.w;
        reinterpret_cast<uint2*>(dst)[i] = t.u;
    }
}

// ---------------------------------------------------------------------------
// Tiled bf16 GEMM: C[M,N] = A[M,K] * B[K,N], row-major bf16, f32 accum.
// Block tile 128x128, K-step 32, double-buffered LDS.
// A tile staged by the Tensor Data Mover (async, TENSORcnt); B tile staged
// transposed by VALU (TDM cannot transpose). 256 thr = 8 waves; wave tile
// 32x64 = 2x4 WMMA 16x16x32 per K-step.
// ---------------------------------------------------------------------------
template <bool OUT_F32>
__global__ __launch_bounds__(256) void gemm_bf16_kernel(
    const bf16* __restrict__ A, const bf16* __restrict__ Bm,
    float* __restrict__ Cf, bf16* __restrict__ Cb,
    int M, int N, int K) {
    constexpr int BM = 128, BN = 128, BK = 32;
    __shared__ __align__(16) bf16 As[2][BM][BK];
    __shared__ __align__(16) bf16 Bs[2][BN][BK];   // Bs[n][k] = B[k][n]

    const int tid  = threadIdx.x;
    const int lane = tid & 31;
    const int wid  = tid >> 5;
    const int l15  = lane & 15;
    const int hi   = lane >> 4;
    const int wm   = wid & 3;      // 32-row group
    const int wn   = wid >> 2;     // 64-col group
    const int rowBase = blockIdx.y * BM;
    const int colBase = blockIdx.x * BN;

    v8f acc[2][4] = {};

    // B stager: thread t covers k = t>>3, n = (t&7)*16 .. +15
    const int bk = tid >> 3;
    const int bn = (tid & 7) * 16;

    auto issueA = [&](int k0, int buf) {
        if (wid == 0)
            tdm_load_2d_bf16(A + (size_t)rowBase * K + k0,
                             (unsigned)(uintptr_t)&As[buf][0][0],
                             /*tile_d0=*/BK, /*tile_d1=*/BM,
                             /*stride_d0=*/(unsigned)K,
                             /*tensor_d0=*/(unsigned)K, /*tensor_d1=*/(unsigned)M);
    };
    auto stageB = [&](int k0, int buf) {
        union { uint4 q[2]; bf16 e[16]; } t;
        const bf16* src = Bm + (size_t)(k0 + bk) * N + colBase + bn;
        t.q[0] = *reinterpret_cast<const uint4*>(src);
        t.q[1] = *reinterpret_cast<const uint4*>(src + 8);
#pragma unroll
        for (int j = 0; j < 16; ++j) Bs[buf][bn + j][bk] = t.e[j];
    };

    // ---- prologue: stage k0 = 0 into buffer 0 ----
    issueA(0, 0);
    stageB(0, 0);
    if (wid == 0) tdm_wait0();
    __syncthreads();

    const int nsteps = K / BK;
    for (int s = 0; s < nsteps; ++s) {
        const int buf = s & 1;
        if (s + 1 < nsteps) {               // kick off next tile while computing
            issueA((s + 1) * BK, buf ^ 1);
            stageB((s + 1) * BK, buf ^ 1);
        }

        FragA a[2], b[4];
#pragma unroll
        for (int im = 0; im < 2; ++im) {
            const bf16* ap = &As[buf][wm * 32 + im * 16 + l15][0];
            a[im].q[0] = *reinterpret_cast<const uint4*>(ap + hi * 16);
            a[im].q[1] = *reinterpret_cast<const uint4*>(ap + hi * 16 + 8);
        }
#pragma unroll
        for (int in = 0; in < 4; ++in) {
            const uint4* bp =
                reinterpret_cast<const uint4*>(&Bs[buf][wn * 64 + in * 16 + l15][hi * 16]);
            b[in].q[0] = bp[0];
            b[in].q[1] = bp[1];
        }
#pragma unroll
        for (int im = 0; im < 2; ++im)
#pragma unroll
            for (int in = 0; in < 4; ++in)
                acc[im][in] = __builtin_amdgcn_wmma_f32_16x16x32_bf16(
                    false, a[im].v, false, b[in].v, (short)0, acc[im][in], false, false);

        if (wid == 0 && s + 1 < nsteps) tdm_wait0();
        __syncthreads();
    }

#pragma unroll
    for (int im = 0; im < 2; ++im)
#pragma unroll
        for (int in = 0; in < 4; ++in)
#pragma unroll
            for (int r = 0; r < 8; ++r) {
                size_t row = rowBase + wm * 32 + im * 16 + r + 8 * hi;
                size_t col = colBase + wn * 64 + in * 16 + l15;
                if (OUT_F32) Cf[row * N + col] = acc[im][in][r];
                else         Cb[row * N + col] = (bf16)acc[im][in][r];
            }
}

// ---------------------------------------------------------------------------
// Flash attention forward (causal). qkv is [B*T, 3*D] bf16:
// Q at col h*64, K at col D + h*64, V at col 2D + h*64.
// Grid: (T/64, B*H); 128 threads = 4 waves; each wave owns 16 query rows.
// ---------------------------------------------------------------------------
__global__ __launch_bounds__(128) void attn_kernel(const bf16* __restrict__ qkv,
                                                   bf16* __restrict__ out) {
    constexpr int T = 2048, D = 1024;
    __shared__ __align__(16) bf16 Vt[64][32];     // [dk][key] transposed V tile
    __shared__ __align__(16) bf16 Ps[4][16][32];  // per-wave P tile [q][key]

    const int tid  = threadIdx.x;
    const int lane = tid & 31;
    const int w    = tid >> 5;
    const int l15  = lane & 15;
    const int hi   = lane >> 4;
    const int bh   = blockIdx.y;
    const int b    = bh >> 4;        // H == 16
    const int h    = bh & 15;
    const int wgQ  = blockIdx.x * 64;
    const int qBase = wgQ + w * 16;
    const size_t rowStride = 3 * D;
    const size_t baseRow   = (size_t)b * T;
    const int qCol = h * 64;
    const int kCol = D + h * 64;
    const int vCol = 2 * D + h * 64;

    // Q fragments (A layout), dk split into two 32-slices
    FragA qf[2];
#pragma unroll
    for (int s = 0; s < 2; ++s) {
        const bf16* qp = qkv + (baseRow + qBase + l15) * rowStride + qCol + 32 * s + hi * 16;
        qf[s].q[0] = *reinterpret_cast<const uint4*>(qp);
        qf[s].q[1] = *reinterpret_cast<const uint4*>(qp + 8);
    }

    float m[8], l[8];
    v8f O[4] = {};
#pragma unroll
    for (int r = 0; r < 8; ++r) { m[r] = -3.0e38f; l[r] = 0.0f; }

    const int nkb = wgQ / 32 + 2;    // key blocks covering keys 0..wgQ+63
    for (int kb = 0; kb < nkb * 32; kb += 32) {
        // cooperatively stage V tile transposed: Vt[dk][key]
        {
            const int k    = tid >> 2;          // key 0..31
            const int dOff = (tid & 3) * 16;    // dk offset
            union { uint4 q[2]; bf16 e[16]; } t;
            const bf16* vp = qkv + (baseRow + kb + k) * rowStride + vCol + dOff;
            t.q[0] = *reinterpret_cast<const uint4*>(vp);
            t.q[1] = *reinterpret_cast<const uint4*>(vp + 8);
#pragma unroll
            for (int j = 0; j < 16; ++j) Vt[dOff + j][k] = t.e[j];
        }
        __syncthreads();

        // scores for two 16-key tiles: S = Q * K^T
        float p[2][8];
#pragma unroll
        for (int t = 0; t < 2; ++t) {
            v8f S = {};
#pragma unroll
            for (int s = 0; s < 2; ++s) {
                FragA kf;   // B layout: elem i -> K[key kBase+l15][dk 32s+16hi+i]
                const bf16* kp =
                    qkv + (baseRow + kb + 16 * t + l15) * rowStride + kCol + 32 * s + 16 * hi;
                kf.q[0] = *reinterpret_cast<const uint4*>(kp);
                kf.q[1] = *reinterpret_cast<const uint4*>(kp + 8);
                S = __builtin_amdgcn_wmma_f32_16x16x32_bf16(
                    false, qf[s].v, false, kf.v, (short)0, S, false, false);
            }
#pragma unroll
            for (int r = 0; r < 8; ++r) {
                int q   = qBase + r + 8 * hi;
                int key = kb + 16 * t + l15;
                p[t][r] = (key <= q) ? S[r] * 0.125f : -3.0e38f;  // 1/sqrt(64)
            }
        }

        // online softmax (row = r + 8*hi lives in the 16-lane half group)
        float alpha[8];
#pragma unroll
        for (int r = 0; r < 8; ++r) {
            float rm = fmaxf(p[0][r], p[1][r]);
            rm = fmaxf(rm, __shfl_xor(rm, 1, 32));
            rm = fmaxf(rm, __shfl_xor(rm, 2, 32));
            rm = fmaxf(rm, __shfl_xor(rm, 4, 32));
            rm = fmaxf(rm, __shfl_xor(rm, 8, 32));
            float mn = fmaxf(m[r], rm);
            alpha[r] = __expf(m[r] - mn);
            float p0 = __expf(p[0][r] - mn);
            float p1 = __expf(p[1][r] - mn);
            float rs = p0 + p1;
            rs += __shfl_xor(rs, 1, 32);
            rs += __shfl_xor(rs, 2, 32);
            rs += __shfl_xor(rs, 4, 32);
            rs += __shfl_xor(rs, 8, 32);
            l[r] = l[r] * alpha[r] + rs;
            m[r] = mn;
            Ps[w][r + 8 * hi][l15]      = (bf16)p0;
            Ps[w][r + 8 * hi][16 + l15] = (bf16)p1;
        }

        // rescale O accumulators
#pragma unroll
        for (int d = 0; d < 4; ++d)
#pragma unroll
            for (int r = 0; r < 8; ++r) O[d][r] *= alpha[r];

        // reload P as an A-layout fragment (per-wave LDS; DS ops are in-order)
        FragA pf;
        pf.q[0] = *reinterpret_cast<const uint4*>(&Ps[w][l15][hi * 16]);
        pf.q[1] = *reinterpret_cast<const uint4*>(&Ps[w][l15][hi * 16 + 8]);

        // O += P * V
#pragma unroll
        for (int d = 0; d < 4; ++d) {
            FragA vf;   // B layout: elem i -> V[key kb+16hi+i][dk d*16+l15]
            const uint4* vp = reinterpret_cast<const uint4*>(&Vt[d * 16 + l15][hi * 16]);
            vf.q[0] = vp[0];
            vf.q[1] = vp[1];
            O[d] = __builtin_amdgcn_wmma_f32_16x16x32_bf16(
                false, pf.v, false, vf.v, (short)0, O[d], false, false);
        }
        __syncthreads();   // before next Vt overwrite
    }

    // normalize and store bf16 attention output [B*T, D]
#pragma unroll
    for (int d = 0; d < 4; ++d)
#pragma unroll
        for (int r = 0; r < 8; ++r) {
            float o = O[d][r] / l[r];
            out[(baseRow + qBase + r + 8 * hi) * D + h * 64 + d * 16 + l15] = (bf16)o;
        }
}

// ---------------------------------------------------------------------------
// Host-side launcher
// ---------------------------------------------------------------------------
extern "C" void kernel_launch(void* const* d_in, const int* in_sizes, int n_in,
                              void* d_out, int out_size, void* d_ws, size_t ws_size,
                              hipStream_t stream) {
    constexpr int B = 4, T = 2048, D = 1024;
    constexpr int M  = B * T;          // 8192
    constexpr int N1 = 3 * D;          // 3072

    const float* x     = (const float*)d_in[0];
    // d_in[1] is the causal tril mask; causality is computed analytically.
    const float* Wqkv  = (const float*)d_in[2];
    const float* Wproj = (const float*)d_in[3];
    float* out = (float*)d_out;

    char* ws = (char*)d_ws;
    bf16* xb     = (bf16*)(ws);                       // 16 MiB
    bf16* wqkvb  = (bf16*)(ws + 16777216);            //  6 MiB
    bf16* wprojb = (bf16*)(ws + 23068672);            //  2 MiB
    bf16* qkvb   = (bf16*)(ws + 25165824);            // 48 MiB
    bf16* attnb  = (bf16*)(ws + 75497472);            // 16 MiB

    const int nx = M * D / 4, nq = D * N1 / 4, np = D * D / 4;
    cvt_f32_to_bf16<<<(nx + 255) / 256, 256, 0, stream>>>(x, xb, nx);
    cvt_f32_to_bf16<<<(nq + 255) / 256, 256, 0, stream>>>(Wqkv, wqkvb, nq);
    cvt_f32_to_bf16<<<(np + 255) / 256, 256, 0, stream>>>(Wproj, wprojb, np);

    // qkv = x @ W_qkv   [8192,1024] x [1024,3072] -> bf16
    gemm_bf16_kernel<false><<<dim3(N1 / 128, M / 128), 256, 0, stream>>>(
        xb, wqkvb, nullptr, qkvb, M, N1, D);

    // flash attention -> attnb bf16 [8192,1024]
    attn_kernel<<<dim3(T / 64, B * 16), 128, 0, stream>>>(qkvb, attnb);

    // out = attn @ W_proj   [8192,1024] x [1024,1024] -> f32
    gemm_bf16_kernel<true><<<dim3(D / 128, M / 128), 256, 0, stream>>>(
        attnb, wprojb, out, nullptr, M, D, D);
}
